// NeuroLex_model_541165879474
// MI455X (gfx1250) — compile-verified, matrix operand order checked by reading, not run on hardware
//
#include <hip/hip_runtime.h>
#include <hip/hip_bf16.h>

typedef __attribute__((ext_vector_type(16))) __bf16 v16b;
typedef __attribute__((ext_vector_type(8)))  __bf16 v8b;
typedef __attribute__((ext_vector_type(4)))  __bf16 v4b;
typedef __attribute__((ext_vector_type(8)))  float  v8f;

#define LDSS 40   // bf16 elems per LDS tile row (32 data + 8 pad) -> 80B rows, 16B aligned

__device__ __forceinline__ float act_apply(float x, int act){
  if(act==1) return 0.5f*x*(1.0f+erff(x*0.70710678118654752440f)); // exact GELU
  if(act==2) return fmaxf(x, 0.0f);                                // ReLU
  return x;
}

// C[M,N] = act(A[M,K] @ B + bias), fp32 in/out, bf16 WMMA f32-accum.
// BT=true : B supplied as W[N,K] row-major (K contiguous)  -> C = A @ W^T
// BT=false: B supplied as B[K,N] row-major (N contiguous)  -> C = A @ B
// Batched via blockIdx.z = b*batchH + h with element strides.
// Requires: K % 4 == 0, 16B-aligned A/B bases, lda/ldb % 4 == 0 (true for all call sites).
template<bool BT>
__global__ __launch_bounds__(256)
void gemm_wmma(const float* __restrict__ A, const float* __restrict__ Bm,
               const float* __restrict__ bias, float* __restrict__ C,
               int M, int N, int K, int lda, int ldb, int ldc, int act,
               int batchH, long sAb, long sAh, long sBb, long sBh, long sCb, long sCh)
{
  __shared__ __align__(16) __bf16 As[128*LDSS];
  __shared__ __align__(16) __bf16 Bs[128*LDSS];

  const int tid  = threadIdx.x;
  const int lane = tid & 31;
  const int wave = tid >> 5;
  const int wm = wave & 1;      // 2 waves along M (64 rows each)
  const int wn = wave >> 1;     // 4 waves along N (32 cols each)
  const int m0 = blockIdx.y * 128;
  const int n0 = blockIdx.x * 128;
  const int zb = (int)blockIdx.z / batchH;
  const int zh = (int)blockIdx.z % batchH;
  A  += (long)zb*sAb + (long)zh*sAh;
  Bm += (long)zb*sBb + (long)zh*sBh;
  C  += (long)zb*sCb + (long)zh*sCh;

  const int lm = lane & 15;
  const int kb = (lane >> 4) * 8;   // CDNA5 A/B frag K-split: 0 or 8

  // staging geometry: chunk q = tid + i*256 -> row r = q>>3, col c = (q&7)*4
  const int srow = tid >> 3;          // + i*32
  const int scol = (tid & 7) * 4;

  v8f acc[4][2];
  v8f zf = {0.f,0.f,0.f,0.f,0.f,0.f,0.f,0.f};
  #pragma unroll
  for(int i=0;i<4;i++){ acc[i][0]=zf; acc[i][1]=zf; }

  float4 ra[4];        // A register stage (4 chunks of 4 floats)
  float4 rbv[4];       // B register stage (BT path)
  float  rbs[16];      // B register stage (non-BT path)

  auto loadA = [&](int k0){
    #pragma unroll
    for(int i=0;i<4;i++){
      int gm = m0 + srow + i*32, gk = k0 + scol;
      bool ok = (gm < M) && (gk < K);
      long idx = ok ? ((long)gm*lda + gk) : 0;
      float4 v = *(const float4*)(A + idx);     // unconditional, clamped address
      ra[i].x = ok ? v.x : 0.f; ra[i].y = ok ? v.y : 0.f;
      ra[i].z = ok ? v.z : 0.f; ra[i].w = ok ? v.w : 0.f;
    }
  };
  auto loadB = [&](int k0){
    if(BT){
      #pragma unroll
      for(int i=0;i<4;i++){
        int gn = n0 + srow + i*32, gk = k0 + scol;
        bool ok = (gn < N) && (gk < K);
        long idx = ok ? ((long)gn*ldb + gk) : 0;
        float4 v = *(const float4*)(Bm + idx);
        rbv[i].x = ok ? v.x : 0.f; rbv[i].y = ok ? v.y : 0.f;
        rbv[i].z = ok ? v.z : 0.f; rbv[i].w = ok ? v.w : 0.f;
      }
    } else {
      #pragma unroll
      for(int i=0;i<16;i++){
        int e = tid + i*256;
        int kk = e >> 7, nn = e & 127;
        int gn = n0 + nn, gk = k0 + kk;
        bool ok = (gn < N) && (gk < K);
        long idx = ok ? ((long)gk*ldb + gn) : 0;
        float v = Bm[idx];
        rbs[i] = ok ? v : 0.f;
      }
    }
  };
  auto storeA = [&](){
    #pragma unroll
    for(int i=0;i<4;i++){
      v4b t = {(__bf16)ra[i].x,(__bf16)ra[i].y,(__bf16)ra[i].z,(__bf16)ra[i].w};
      *(v4b*)(&As[(srow + i*32)*LDSS + scol]) = t;   // one ds_store_b64
    }
  };
  auto storeB = [&](){
    if(BT){
      #pragma unroll
      for(int i=0;i<4;i++){
        v4b t = {(__bf16)rbv[i].x,(__bf16)rbv[i].y,(__bf16)rbv[i].z,(__bf16)rbv[i].w};
        *(v4b*)(&Bs[(srow + i*32)*LDSS + scol]) = t;
      }
    } else {
      #pragma unroll
      for(int i=0;i<16;i++){
        int e = tid + i*256;
        int kk = e >> 7, nn = e & 127;
        Bs[nn*LDSS + kk] = (__bf16)rbs[i];
      }
    }
  };

  loadA(0); loadB(0);

  for(int k0=0; k0<K; k0+=32){
    storeA(); storeB();
    __syncthreads();

    if(k0 + 32 < K){                 // uniform branch: prefetch next tile into regs
      loadA(k0 + 32);
      loadB(k0 + 32);
      if(k0 + 64 < K){               // L2 prefetch two tiles ahead (global_prefetch_b8)
        int gm = m0 + srow, gk = k0 + 64 + scol;
        if(gm < M && gk < K) __builtin_prefetch(&A[(long)gm*lda + gk], 0, 1);
      }
    }

    // ---- fragment loads: two ds_load_b128 per frag, per documented layout ----
    v16b af[4], bfr[2];
    #pragma unroll
    for(int i=0;i<4;i++){
      int row = wm*64 + i*16 + lm;
      v8b lo = *(const v8b*)(&As[row*LDSS + kb]);
      v8b hi = *(const v8b*)(&As[row*LDSS + 16 + kb]);
      af[i] = __builtin_shufflevector(lo,hi,0,1,2,3,4,5,6,7,8,9,10,11,12,13,14,15);
    }
    #pragma unroll
    for(int j=0;j<2;j++){
      int col = wn*32 + j*16 + lm;
      v8b lo = *(const v8b*)(&Bs[col*LDSS + kb]);
      v8b hi = *(const v8b*)(&Bs[col*LDSS + 16 + kb]);
      bfr[j] = __builtin_shufflevector(lo,hi,0,1,2,3,4,5,6,7,8,9,10,11,12,13,14,15);
    }
    #pragma unroll
    for(int i=0;i<4;i++)
      #pragma unroll
      for(int j=0;j<2;j++)
        acc[i][j] = __builtin_amdgcn_wmma_f32_16x16x32_bf16(
            false, af[i], false, bfr[j], (short)0, acc[i][j], false, false);
    __syncthreads();
  }

  // ---- epilogue: C/D layout: lanes<16 rows 0..7, lanes>=16 rows 8..15 ----
  const int hi8 = (lane >> 4) * 8;
  const bool full = (m0 + 128 <= M) && (n0 + 128 <= N);
  if(full){
    #pragma unroll
    for(int j=0;j<2;j++){
      int colg = n0 + wn*32 + j*16 + lm;
      float bv = (bias != nullptr) ? bias[colg] : 0.0f;
      #pragma unroll
      for(int i=0;i<4;i++){
        long rbase = (long)(m0 + wm*64 + i*16 + hi8)*ldc + colg;
        #pragma unroll
        for(int v=0;v<8;v++)
          C[rbase + (long)v*ldc] = act_apply(acc[i][j][v] + bv, act);
      }
    }
  } else {
    #pragma unroll
    for(int j=0;j<2;j++){
      int colg = n0 + wn*32 + j*16 + lm;
      float bv = (bias != nullptr && colg < N) ? bias[colg] : 0.0f;
      #pragma unroll
      for(int i=0;i<4;i++){
        #pragma unroll
        for(int v=0;v<8;v++){
          int rowg = m0 + wm*64 + i*16 + hi8 + v;
          if(rowg < M && colg < N)
            C[(long)rowg*ldc + colg] = act_apply(acc[i][j][v] + bv, act);
        }
      }
    }
  }
}

// ---------------- support kernels ----------------

// grid: (ceil(NN/256), B) ; y[b*NN+i] = x[b*NN+i] + pos[i]
__global__ void k_addpos(const float* __restrict__ x, const float* __restrict__ pos,
                         float* __restrict__ y, int NN){
  int i = blockIdx.x*256 + threadIdx.x;
  if(i < NN){
    long o = (long)blockIdx.y*NN + i;
    y[o] = x[o] + pos[i];
  }
}

// Y = LayerNorm(X + R) * w + b   (R may be null; Y may alias X)
__global__ __launch_bounds__(256)
void k_ln(const float* __restrict__ X, const float* __restrict__ R,
          const float* __restrict__ w, const float* __restrict__ bb,
          float* __restrict__ Y, int D){
  long row = blockIdx.x;
  const float* xp = X + row*(long)D;
  const float* rp = R ? R + row*(long)D : nullptr;
  float* yp = Y + row*(long)D;
  __shared__ float s1[256], s2[256];
  float s=0.f, sq=0.f;
  for(int i=threadIdx.x;i<D;i+=256){
    float t = xp[i] + (rp ? rp[i] : 0.0f);
    s += t; sq += t*t;
  }
  s1[threadIdx.x]=s; s2[threadIdx.x]=sq; __syncthreads();
  for(int o=128;o>0;o>>=1){
    if(threadIdx.x<o){ s1[threadIdx.x]+=s1[threadIdx.x+o]; s2[threadIdx.x]+=s2[threadIdx.x+o]; }
    __syncthreads();
  }
  float mean = s1[0]/(float)D;
  float var  = s2[0]/(float)D - mean*mean;
  float inv  = rsqrtf(var + 1e-5f);
  for(int i=threadIdx.x;i<D;i+=256){
    float t = xp[i] + (rp ? rp[i] : 0.0f);
    yp[i] = (t-mean)*inv*w[i] + bb[i];
  }
}

__global__ __launch_bounds__(256)
void k_softmax(float* __restrict__ X, int S, float scale){
  long row = blockIdx.x;
  float* x = X + row*(long)S;
  __shared__ float sr[256];
  float m = -3.4e38f;
  for(int i=threadIdx.x;i<S;i+=256) m = fmaxf(m, x[i]*scale);
  sr[threadIdx.x]=m; __syncthreads();
  for(int o=128;o>0;o>>=1){ if(threadIdx.x<o) sr[threadIdx.x]=fmaxf(sr[threadIdx.x],sr[threadIdx.x+o]); __syncthreads(); }
  m = sr[0]; __syncthreads();
  float s=0.f;
  for(int i=threadIdx.x;i<S;i+=256){ float e = expf(x[i]*scale - m); x[i]=e; s+=e; }
  sr[threadIdx.x]=s; __syncthreads();
  for(int o=128;o>0;o>>=1){ if(threadIdx.x<o) sr[threadIdx.x]+=sr[threadIdx.x+o]; __syncthreads(); }
  float inv = 1.0f/sr[0];
  for(int i=threadIdx.x;i<S;i+=256) x[i]*=inv;
}

__global__ __launch_bounds__(256)
void k_rowsq(const float* __restrict__ E, float* __restrict__ e2, int L){
  int k = blockIdx.x;
  __shared__ float sr[256];
  float s=0.f;
  for(int i=threadIdx.x;i<L;i+=256){ float t=E[(long)k*L+i]; s+=t*t; }
  sr[threadIdx.x]=s; __syncthreads();
  for(int o=128;o>0;o>>=1){ if(threadIdx.x<o) sr[threadIdx.x]+=sr[threadIdx.x+o]; __syncthreads(); }
  if(threadIdx.x==0) e2[k]=sr[0];
}

__global__ __launch_bounds__(256)
void k_vq(const float* __restrict__ z, const float* __restrict__ zE,
          const float* __restrict__ e2, const float* __restrict__ E,
          float* __restrict__ zq, float* __restrict__ idxf,
          int* __restrict__ counts, float* __restrict__ lossAcc, int Kc, int L){
  int b = blockIdx.x;
  __shared__ float sr[256];
  __shared__ float sd[256]; __shared__ int si[256];
  float s=0.f;
  for(int i=threadIdx.x;i<L;i+=256){ float t=z[(long)b*L+i]; s+=t*t; }
  sr[threadIdx.x]=s; __syncthreads();
  for(int o=128;o>0;o>>=1){ if(threadIdx.x<o) sr[threadIdx.x]+=sr[threadIdx.x+o]; __syncthreads(); }
  float z2 = sr[0];
  float best = 3.4e38f; int bi = 0x7fffffff;
  for(int k=threadIdx.x;k<Kc;k+=256){
    float d = z2 + e2[k] - 2.0f*zE[(long)b*Kc+k];
    if(d < best || (d==best && k<bi)){ best=d; bi=k; }
  }
  sd[threadIdx.x]=best; si[threadIdx.x]=bi; __syncthreads();
  for(int o=128;o>0;o>>=1){
    if(threadIdx.x<o){
      float d2=sd[threadIdx.x+o]; int i2=si[threadIdx.x+o];
      if(d2<sd[threadIdx.x] || (d2==sd[threadIdx.x] && i2<si[threadIdx.x])){ sd[threadIdx.x]=d2; si[threadIdx.x]=i2; }
    }
    __syncthreads();
  }
  int idx = si[0];
  if(threadIdx.x==0){ idxf[b]=(float)idx; atomicAdd(&counts[idx],1); }
  float ls=0.f;
  for(int i=threadIdx.x;i<L;i+=256){
    float q = E[(long)idx*L+i];
    zq[(long)b*L+i]=q;
    float d = q - z[(long)b*L+i];
    ls += d*d;
  }
  __syncthreads();
  sr[threadIdx.x]=ls; __syncthreads();
  for(int o=128;o>0;o>>=1){ if(threadIdx.x<o) sr[threadIdx.x]+=sr[threadIdx.x+o]; __syncthreads(); }
  if(threadIdx.x==0) atomicAdd(lossAcc, sr[0]);
}

__global__ __launch_bounds__(256)
void k_vqfin(const int* __restrict__ counts, const float* __restrict__ lossAcc,
             float* __restrict__ outLoss, float* __restrict__ outPerp,
             int Kc, int Bsz, int L, float beta){
  __shared__ float sr[256];
  float s=0.f;
  for(int k=threadIdx.x;k<Kc;k+=256){
    float avg = (float)counts[k]/(float)Bsz;
    s += avg * logf(avg + 1e-10f);
  }
  sr[threadIdx.x]=s; __syncthreads();
  for(int o=128;o>0;o>>=1){ if(threadIdx.x<o) sr[threadIdx.x]+=sr[threadIdx.x+o]; __syncthreads(); }
  if(threadIdx.x==0){
    *outPerp = expf(-sr[0]);
    *outLoss = beta * (*lossAcc) / ((float)Bsz * (float)L);
  }
}

// grid: (N, B) ; one block per output row: out[b,i,j] = 0.5*(r[b,i,j]+r[b,j,i])
__global__ void k_sym(const float* __restrict__ r, float* __restrict__ o, int Nn){
  int ii = blockIdx.x;
  int b  = blockIdx.y;
  int j  = threadIdx.x;
  if(j < Nn){
    long base = (long)b*Nn*Nn;
    o[base + (long)ii*Nn + j] = 0.5f*(r[base + (long)ii*Nn + j] + r[base + (long)j*Nn + ii]);
  }
}

// ---------------- host orchestration ----------------

extern "C" void kernel_launch(void* const* d_in, const int* in_sizes, int n_in,
                              void* d_out, int out_size, void* d_ws, size_t ws_size,
                              hipStream_t stream)
{
  (void)in_sizes; (void)n_in; (void)out_size; (void)ws_size;
  const float* x = (const float*)d_in[0];
  auto P = [&](int i){ return (const float*)d_in[i]; };

  const int B=256, N=200, HE=4, HD=2, LAT=256, Kc=512;
  const long Mtok = (long)B*N;        // 51200
  const int  NN   = N*N;              // 40000
  const long RECON = (long)B*N*N;     // 10,240,000

  float* ws = (float*)d_ws;
  size_t off=0;
  auto take=[&](size_t n){ float* p = ws + off; off += n; return p; };
  float* wsH   = take((size_t)Mtok*512);     // main activation [B*N, 512]
  float* wsBig = take((size_t)Mtok*1536);    // qkv / ff1 / misc
  float* wsScr = take((size_t)B*HE*N*N);     // img / attention scores
  float* wsT   = take((size_t)Mtok*512);     // attn-out / ff2 / r
  float* ws32a = take((size_t)Mtok*32);
  float* ws32b = take((size_t)Mtok*32);
  float* vz    = take((size_t)B*LAT);
  float* vgt   = take((size_t)B*512);
  float* vzE   = take((size_t)B*Kc);
  float* vzq   = take((size_t)B*LAT);
  float* ve2   = take((size_t)Kc);
  float* vloss = take(1);
  int*   vcnt  = (int*)take(Kc);

  float* out     = (float*)d_out;
  float* outLoss = out + RECON;
  float* outPerp = out + RECON + 1;
  float* outIdx  = out + RECON + 2;

  auto gemm = [&](const float* A, const float* W, const float* bias, float* C,
                  long M, int Nc, int K, int lda, int ldb, int ldc, int act){
    dim3 g((unsigned)((Nc+127)/128), (unsigned)((M+127)/128), 1);
    gemm_wmma<true><<<g, dim3(256), 0, stream>>>(A,W,bias,C,(int)M,Nc,K,lda,ldb,ldc,act,
                                                 1,0,0,0,0,0,0);
  };
  auto gemmT_b = [&](const float* A, const float* W, float* C,
                     int M,int Nc,int K,int lda,int ldb,int ldc,int Bct,int H,
                     long sAb,long sAh,long sBb,long sBh,long sCb,long sCh){
    dim3 g((unsigned)((Nc+127)/128),(unsigned)((M+127)/128),(unsigned)(Bct*H));
    gemm_wmma<true><<<g,dim3(256),0,stream>>>(A,W,nullptr,C,M,Nc,K,lda,ldb,ldc,0,
                                              H,sAb,sAh,sBb,sBh,sCb,sCh);
  };
  auto gemmN_b = [&](const float* A, const float* Bm, float* C,
                     int M,int Nc,int K,int lda,int ldb,int ldc,int Bct,int H,
                     long sAb,long sAh,long sBb,long sBh,long sCb,long sCh){
    dim3 g((unsigned)((Nc+127)/128),(unsigned)((M+127)/128),(unsigned)(Bct*H));
    gemm_wmma<false><<<g,dim3(256),0,stream>>>(A,Bm,nullptr,C,M,Nc,K,lda,ldb,ldc,0,
                                               H,sAb,sAh,sBb,sBh,sCb,sCh);
  };
  auto ln = [&](const float* X, const float* R, const float* w, const float* b,
                float* Y, long rows, int D){
    k_ln<<<(unsigned)rows,256,0,stream>>>(X,R,w,b,Y,D);
  };

  // ---------------- BNTF encoder front ----------------
  k_addpos<<<dim3((NN+255)/256, B), 256, 0, stream>>>(x, P(1), wsScr, NN);
  gemm(wsScr, P(2), P(3), ws32a, Mtok, 32, 200, 200, 200, 32, 1);     // cf.l1 + GELU
  ln(ws32a, nullptr, P(4), P(5), ws32a, Mtok, 32);                    // cf.ln
  gemm(ws32a, P(6), P(7), ws32b, Mtok, 32, 32, 32,32,32, 1);          // rb.l1 + GELU
  gemm(ws32b, P(8), P(9), wsT,   Mtok, 32, 32, 32,32,32, 0);          // rb.l2
  ln(ws32a, wsT, P(10), P(11), ws32a, Mtok, 32);                      // rb.ln (residual)
  gemm(ws32a, P(12), P(13), wsH, Mtok, 512, 32, 32,32,512, 0);        // cf.l2 -> [B,N,512]

  // ---------------- transformer layer (shared enc/dec) ----------------
  auto tlayer = [&](int p0, int H, int act){
    const int D=512, FF=1024, S=200;
    const int dh = D/H;
    const float scale = 1.0f/sqrtf((float)dh);
    // qkv
    gemm(wsH, P(p0+0), P(p0+1), wsBig, Mtok, 3*D, D, D, D, 3*D, 0);
    // scores = Q @ K^T  (batched over B*H)
    gemmT_b(wsBig, wsBig + D, wsScr, S, S, dh, 3*D, 3*D, S, B, H,
            (long)S*3*D, (long)dh, (long)S*3*D, (long)dh, (long)H*S*S, (long)S*S);
    k_softmax<<<(unsigned)(B*H*S),256,0,stream>>>(wsScr, S, scale);
    // ctx = P @ V  -> wsT as [B,S,D] with per-head column offset
    gemmN_b(wsScr, wsBig + 2*D, wsT, S, dh, S, S, 3*D, D, B, H,
            (long)H*S*S, (long)S*S, (long)S*3*D, (long)dh, (long)S*D, (long)dh);
    // out proj + residual LN
    gemm(wsT, P(p0+2), P(p0+3), wsBig, Mtok, D, D, D, D, D, 0);
    ln(wsH, wsBig, P(p0+4), P(p0+5), wsH, Mtok, D);
    // FF
    gemm(wsH, P(p0+6), P(p0+7), wsBig, Mtok, FF, D, D, D, FF, act);
    gemm(wsBig, P(p0+8), P(p0+9), wsT, Mtok, D, FF, FF, FF, D, 0);
    ln(wsH, wsT, P(p0+10), P(p0+11), wsH, Mtok, D);
  };

  for(int i=0;i<4;i++) tlayer(14+12*i, HE, 1);                        // encoder (GELU)

  // ---------------- encoder tail + VQ ----------------
  ln(wsH, nullptr, P(62), P(63), wsH, Mtok, 512);                     // enc_norm
  gemm(wsH, P(64), P(65), ws32a, Mtok, 4, 512, 512, 512, 4, 1);       // dimred + GELU
  gemm(ws32a, P(66), P(67), vgt, B, 512, 800, 800, 800, 512, 1);      // g_enc.l1 + GELU
  gemm(vgt, P(68), P(69), vz, B, 256, 512, 512, 512, 256, 0);         // g_enc.l2 -> z
  k_rowsq<<<(unsigned)Kc,256,0,stream>>>(P(70), ve2, LAT);            // |E_k|^2
  gemm(vz, P(70), nullptr, vzE, B, Kc, LAT, LAT, LAT, Kc, 0);         // z @ E^T
  hipMemsetAsync(vloss, 0, (size_t)(1+Kc)*sizeof(float), stream);     // loss + counts
  k_vq<<<(unsigned)B,256,0,stream>>>(vz, vzE, ve2, P(70), vzq, outIdx, vcnt, vloss, Kc, LAT);
  k_vqfin<<<1,256,0,stream>>>(vcnt, vloss, outLoss, outPerp, Kc, B, LAT, 0.5f);

  // ---------------- decoder ----------------
  gemm(vzq, P(71), P(72), wsBig, B, 1600, 256, 256, 256, 1600, 1);    // proj.l1 + GELU
  gemm(wsBig, P(73), P(74), ws32a, B, 800, 1600, 1600, 1600, 800, 0); // proj.l2 -> pr
  gemm(ws32a, P(75), P(76), ws32b, Mtok, 32, 4, 4, 4, 32, 2);         // fus_l1 + ReLU
  ln(ws32b, nullptr, P(77), P(78), ws32a, Mtok, 32);                  // fus_ln
  gemm(ws32a, P(79), P(80), ws32b, Mtok, 32, 32, 32,32,32, 1);        // fus_rb.l1 + GELU
  gemm(ws32b, P(81), P(82), wsT,   Mtok, 32, 32, 32,32,32, 0);        // fus_rb.l2
  ln(ws32a, wsT, P(83), P(84), ws32a, Mtok, 32);                      // fus_rb.ln
  gemm(ws32a, P(85), P(86), wsH, Mtok, 512, 32, 32,32,512, 0);        // fus_l2

  for(int i=0;i<2;i++) tlayer(87+12*i, HD, 2);                        // decoder (ReLU)

  ln(wsH, nullptr, P(111), P(112), wsH, Mtok, 512);                   // dec norm
  gemm(wsH, P(113), P(114), wsBig, Mtok, 400, 512, 512,512,400, 1);   // g.l1 + GELU
  gemm(wsBig, P(115), P(116), wsT, Mtok, 200, 400, 400,400,200, 0);   // g.l2 -> r
  k_sym<<<dim3(N, B), 256, 0, stream>>>(wsT, out, N);                 // recon = (r+r^T)/2
}
